// get_model_13864154431842
// MI455X (gfx1250) — compile-verified, matrix operand order checked by reading, not run on hardware
//
#include <hip/hip_runtime.h>
#include <cmath>

#define BATCH 8
#define NPTS  2048
#define KNN   20

typedef __bf16 bf16_t;
typedef __attribute__((ext_vector_type(16))) __bf16 v16bf;
typedef __attribute__((ext_vector_type(8)))  float  v8f;
typedef int v4i_t __attribute__((vector_size(16)));

// ---------------------------------------------------------------------------
// Async global->LDS copies (gfx1250, ASYNCcnt path), with safe fallback
// ---------------------------------------------------------------------------
#if defined(__HIP_DEVICE_COMPILE__) && __has_builtin(__builtin_amdgcn_global_load_async_to_lds_b128)
#define ASYNC_LDS 1
#else
#define ASYNC_LDS 0
#endif

__device__ __forceinline__ void async_b128(const void* g, void* l) {
#if ASYNC_LDS
  typedef __attribute__((address_space(1))) v4i_t* gp_t;
  typedef __attribute__((address_space(3))) v4i_t* lp_t;
  __builtin_amdgcn_global_load_async_to_lds_b128((gp_t)g, (lp_t)l, 0, 0);
#else
  __builtin_memcpy(l, g, 16);
#endif
}

template <int N>
__device__ __forceinline__ void wait_async() {
#if ASYNC_LDS
#if __has_builtin(__builtin_amdgcn_s_wait_asynccnt)
  __builtin_amdgcn_s_wait_asynccnt(N);
#else
  asm volatile("s_wait_asynccnt %0" ::"i"(N) : "memory");
#endif
#endif
}

// ---------------------------------------------------------------------------
// WMMA helpers (gfx1250, wave32). D = A(16x32 bf16) * B(32x16 bf16) + C(f32)
// ---------------------------------------------------------------------------
__device__ __forceinline__ v8f wmma_bf16(v16bf a, v16bf b, v8f c) {
  return __builtin_amdgcn_wmma_f32_16x16x32_bf16(
      /*neg_a=*/false, a, /*neg_b=*/false, b,
      /*c_mod=*/(short)0, c, /*reuse_a=*/false, /*reuse_b=*/false);
}

// K-index striping for 16-bit A/B fragments (ISA 7.12.2)
__device__ __forceinline__ int frag_k(int v, int g) {
  return (v < 4) ? (2 * v + 8 * g) : (16 + 2 * (v - 4) + 8 * g);
}

__device__ __forceinline__ v16bf load_frag_a_lds(const bf16_t* tile, int ld) {
  const int lane = threadIdx.x & 31;
  const int row = lane & 15, g = lane >> 4;
  v16bf f;
#pragma unroll
  for (int v = 0; v < 8; ++v) {
    const int k = frag_k(v, g);
    f[2 * v]     = tile[row * ld + k];
    f[2 * v + 1] = tile[row * ld + k + 1];
  }
  return f;
}

__device__ __forceinline__ v16bf load_frag_b_lds(const bf16_t* tile, int ld) {
  const int lane = threadIdx.x & 31;
  const int col = lane & 15, g = lane >> 4;
  v16bf f;
#pragma unroll
  for (int v = 0; v < 8; ++v) {
    const int k = frag_k(v, g);
    f[2 * v]     = tile[k * ld + col];
    f[2 * v + 1] = tile[(k + 1) * ld + col];
  }
  return f;
}

// B fragment straight from a global f32 weight matrix, row-guarded (K padding)
__device__ __forceinline__ v16bf load_frag_b_glob(const float* __restrict__ W,
                                                  int ldw, int k0, int krows,
                                                  int col0) {
  const int lane = threadIdx.x & 31;
  const int col = col0 + (lane & 15), g = lane >> 4;
  v16bf f;
#pragma unroll
  for (int v = 0; v < 8; ++v) {
    const int k = k0 + frag_k(v, g);
    f[2 * v]     = (k < krows)     ? (bf16_t)W[(size_t)k * ldw + col]       : (bf16_t)0.f;
    f[2 * v + 1] = (k + 1 < krows) ? (bf16_t)W[(size_t)(k + 1) * ldw + col] : (bf16_t)0.f;
  }
  return f;
}

// ---------------------------------------------------------------------------
// 0) f32 -> bf16 conversion (weights, done once per launch)
// ---------------------------------------------------------------------------
__global__ void cvt_bf16_kernel(const float* __restrict__ src,
                                bf16_t* __restrict__ dst, int n) {
  const int i = blockIdx.x * 256 + threadIdx.x;
  if (i < n) dst[i] = (bf16_t)src[i];
}

// ---------------------------------------------------------------------------
// 1) brute-force kNN (top-20 smallest squared distance, self included)
// ---------------------------------------------------------------------------
template <int C>
__global__ void knn_kernel(const float* __restrict__ feat,  // [B,N,C]
                           int* __restrict__ knn_idx) {     // [B,N,KNN]
  constexpr int TJ = 128;
  const int b = blockIdx.y;
  const int i = blockIdx.x * 128 + threadIdx.x;
  __shared__ float tile[TJ * C];

  const float* fb = feat + (size_t)b * NPTS * C;
  float xi[C];
#pragma unroll
  for (int c = 0; c < C; ++c) xi[c] = fb[(size_t)i * C + c];

  float bd[KNN];
  int   bi[KNN];
#pragma unroll
  for (int k = 0; k < KNN; ++k) { bd[k] = 3.0e38f; bi[k] = 0; }

  for (int j0 = 0; j0 < NPTS; j0 += TJ) {
    __syncthreads();
    for (int t = threadIdx.x; t < TJ * C; t += 128)
      tile[t] = fb[(size_t)j0 * C + t];
    // prefetch next candidate tile (global_prefetch_b8)
    {
      const size_t off = (size_t)(j0 + TJ) * C + (size_t)threadIdx.x * 16;
      if (off < (size_t)NPTS * C) __builtin_prefetch(fb + off, 0, 1);
    }
    __syncthreads();
    for (int jj = 0; jj < TJ; ++jj) {
      float d = 0.f;
#pragma unroll
      for (int c = 0; c < C; ++c) {
        const float t = xi[c] - tile[jj * C + c];
        d = fmaf(t, t, d);
      }
      if (d < bd[KNN - 1]) {
        int p = KNN - 1;
        while (p > 0 && bd[p - 1] > d) {
          bd[p] = bd[p - 1]; bi[p] = bi[p - 1]; --p;
        }
        bd[p] = d; bi[p] = j0 + jj;
      }
    }
  }
  int* out = knn_idx + ((size_t)b * NPTS + i) * KNN;
#pragma unroll
  for (int k = 0; k < KNN; ++k) out[k] = bi[k];
}

// ---------------------------------------------------------------------------
// 2) fused EdgeConv: gather -> relu(E@w1+b1) -> (@w2+b2) -> max over K
//    4 points / workgroup => 80 edge rows = 5 WMMA M-tiles, 4 waves
// ---------------------------------------------------------------------------
template <int C, int OUT, typename OT>
__global__ void edge_conv_kernel(const float* __restrict__ feat,  // [B,N,C]
                                 const int* __restrict__ knn_idx, // [B,N,KNN]
                                 const float* __restrict__ w1,    // [2C,64]
                                 const float* __restrict__ b1,    // [64]
                                 const float* __restrict__ w2,    // [64,OUT]
                                 const float* __restrict__ b2,    // [OUT]
                                 OT* __restrict__ outf) {         // [B,N,OUT]
  constexpr int P   = 4;
  constexpr int M   = P * KNN;              // 80 edge rows
  constexpr int KP  = (2 * C + 31) & ~31;   // edge-feature dim padded to 32
  constexpr int HID = 64;
  constexpr int MT  = M / 16;               // 5 M-tiles

  const int b    = blockIdx.y;
  const int p0   = blockIdx.x * P;
  const int tid  = threadIdx.x;
  const int wave = tid >> 5;
  const int lane = tid & 31;
  const int g    = lane >> 4;

  __shared__ bf16_t E[M * KP];
  __shared__ bf16_t H1[M * HID];
  __shared__ bf16_t H2[M * OUT];

  const float* fb = feat + (size_t)b * NPTS * C;

  for (int t = tid; t < M * KP; t += 128) {
    const int r = t / KP, c = t % KP;
    const int p = p0 + r / KNN;
    const int k = r % KNN;
    float v = 0.f;
    if (c < C) {
      v = fb[(size_t)p * C + c];
    } else if (c < 2 * C) {
      const int j = knn_idx[((size_t)b * NPTS + p) * KNN + k];
      v = fb[(size_t)j * C + (c - C)] - fb[(size_t)p * C + (c - C)];
    }
    E[t] = (bf16_t)v;
  }
  __syncthreads();

  for (int job = wave; job < MT * (HID / 16); job += 4) {
    const int mt = job / (HID / 16), nt = job % (HID / 16);
    v8f acc = {};
#pragma unroll
    for (int kc = 0; kc < KP / 32; ++kc) {
      v16bf a  = load_frag_a_lds(E + mt * 16 * KP + kc * 32, KP);
      v16bf bb = load_frag_b_glob(w1, HID, kc * 32, 2 * C, nt * 16);
      acc = wmma_bf16(a, bb, acc);
    }
    const int nn = nt * 16 + (lane & 15);
    const float bias = b1[nn];
#pragma unroll
    for (int v = 0; v < 8; ++v) {
      const int m = mt * 16 + v + 8 * g;
      float val = acc[v] + bias;
      val = val > 0.f ? val : 0.f;
      H1[m * HID + nn] = (bf16_t)val;
    }
  }
  __syncthreads();

  for (int job = wave; job < MT * (OUT / 16); job += 4) {
    const int mt = job / (OUT / 16), nt = job % (OUT / 16);
    v8f acc = {};
#pragma unroll
    for (int kc = 0; kc < HID / 32; ++kc) {
      v16bf a  = load_frag_a_lds(H1 + mt * 16 * HID + kc * 32, HID);
      v16bf bb = load_frag_b_glob(w2, OUT, kc * 32, HID, nt * 16);
      acc = wmma_bf16(a, bb, acc);
    }
    const int nn = nt * 16 + (lane & 15);
    const float bias = b2[nn];
#pragma unroll
    for (int v = 0; v < 8; ++v) {
      const int m = mt * 16 + v + 8 * g;
      H2[m * OUT + nn] = (bf16_t)(acc[v] + bias);
    }
  }
  __syncthreads();

  for (int t = tid; t < P * OUT; t += 128) {
    const int p = t / OUT, c = t % OUT;
    float mval = -3.0e38f;
    for (int k = 0; k < KNN; ++k)
      mval = fmaxf(mval, (float)H2[(p * KNN + k) * OUT + c]);
    outf[((size_t)b * NPTS + p0 + p) * OUT + c] = (OT)mval;
  }
}

// ---------------------------------------------------------------------------
// 3) global max over N (feat2 is [B,N,128] bf16)
// ---------------------------------------------------------------------------
__global__ void global_max_kernel(const bf16_t* __restrict__ feat,
                                  float* __restrict__ glob) {
  const int b = blockIdx.x, c = threadIdx.x;  // blockDim = 128
  float m = -3.0e38f;
  for (int n = 0; n < NPTS; ++n)
    m = fmaxf(m, (float)feat[((size_t)b * NPTS + n) * 128 + c]);
  glob[b * 128 + c] = m;
}

// ---------------------------------------------------------------------------
// 4) generic 64x64-tile bf16 WMMA GEMM: C = act(A@W + bias)
//    A [M x K] bf16 (async-copied tiles), W [K x realN] bf16, output OT.
// ---------------------------------------------------------------------------
template <int ACT, typename OT>
__global__ void gemm_kernel(const bf16_t* __restrict__ A, int lda,
                            const bf16_t* __restrict__ W, int ldw,
                            const float* __restrict__ bias,
                            OT* __restrict__ Co, int ldc, int col0,
                            int K, int realN, int asyncB) {
  __shared__ bf16_t As[64 * 32];
  __shared__ bf16_t Bs[32 * 64];
  const int tid = threadIdx.x, wave = tid >> 5, lane = tid & 31;
  const int g = lane >> 4;
  const int m0 = blockIdx.x * 64;
  const int n0 = blockIdx.y * 64;

  v8f acc[4] = {v8f{}, v8f{}, v8f{}, v8f{}};
  for (int kc = 0; kc < K; kc += 32) {
    __syncthreads();
    // A tile: 64 rows x 32 bf16 = 64B/row -> 4 x b128 segments per row
    for (int t = tid; t < 256; t += 128) {
      const int r = t >> 2, seg = t & 3;
      async_b128(A + (size_t)(m0 + r) * lda + kc + seg * 8,
                 As + r * 32 + seg * 8);
    }
    if (asyncB) {  // 32 rows x 64 bf16 = 128B/row -> 8 segments per row
      for (int t = tid; t < 256; t += 128) {
        const int r = t >> 3, seg = t & 7;
        async_b128(W + (size_t)(kc + r) * ldw + n0 + seg * 8,
                   Bs + r * 64 + seg * 8);
      }
    } else {
      for (int t = tid; t < 32 * 64; t += 128) {
        const int r = t >> 6, c = t & 63;
        const int col = n0 + c;
        Bs[t] = (col < realN) ? W[(size_t)(kc + r) * ldw + col] : (bf16_t)0.f;
      }
    }
    wait_async<0>();
    __syncthreads();
    v16bf a = load_frag_a_lds(As + wave * 16 * 32, 32);
#pragma unroll
    for (int ns = 0; ns < 4; ++ns) {
      v16bf bb = load_frag_b_lds(Bs + ns * 16, 64);
      acc[ns] = wmma_bf16(a, bb, acc[ns]);
    }
  }
#pragma unroll
  for (int ns = 0; ns < 4; ++ns) {
    const int n = n0 + ns * 16 + (lane & 15);
    if (n < realN) {
      const float bv = bias ? bias[n] : 0.f;
#pragma unroll
      for (int v = 0; v < 8; ++v) {
        const int m = m0 + wave * 16 + v + 8 * g;
        float val = acc[ns][v] + bv;
        if (ACT == 1) val = val > 0.f ? val : 0.f;
        Co[(size_t)m * ldc + col0 + n] = (OT)val;
      }
    }
  }
}

// ---------------------------------------------------------------------------
// 5) flash attention with double-buffered async K/V chunks.
//    qkv [B,N,384] bf16 (q|k|v, 4 heads x 32), out att [B,N,128] bf16
// ---------------------------------------------------------------------------
__device__ __forceinline__ void attn_issue_kv(const bf16_t* __restrict__ base,
                                              int j0, int h, int tid,
                                              bf16_t* Kb, bf16_t* Vb) {
  const int r = tid >> 2, seg = tid & 3;  // 32 rows x 4 x b128, 128 threads
  const size_t row = (size_t)(j0 + r) * 384;
  async_b128(base + row + 128 + h * 32 + seg * 8, Kb + r * 32 + seg * 8);
  async_b128(base + row + 256 + h * 32 + seg * 8, Vb + r * 32 + seg * 8);
}

__global__ void attention_kernel(const bf16_t* __restrict__ qkv,
                                 bf16_t* __restrict__ att) {
  constexpr int DH = 32;
  const int b = blockIdx.z, h = blockIdx.y;
  const int tid = threadIdx.x, wave = tid >> 5, lane = tid & 31;
  const int g = lane >> 4;
  const int n0 = blockIdx.x * 64 + wave * 16;
  const float scale = 0.17677669529663687f;  // 32^-0.5, applied to S

  __shared__ bf16_t Qs[4][16 * DH];
  __shared__ bf16_t Ks[2][32 * DH];
  __shared__ bf16_t Vs[2][32 * DH];
  __shared__ bf16_t Ps[4][16 * 32];

  const bf16_t* base = qkv + (size_t)b * NPTS * 384;

  // async-stage this wave's Q tile (16 rows x 32 bf16 = 2 x b128 per lane)
  for (int t = lane; t < 64; t += 32) {
    const int r = t >> 2, seg = t & 3;
    async_b128(base + (size_t)(n0 + r) * 384 + h * DH + seg * 8,
               Qs[wave] + r * DH + seg * 8);
  }
  attn_issue_kv(base, 0, h, tid, Ks[0], Vs[0]);
  wait_async<0>();
  __syncthreads();
  v16bf qa = load_frag_a_lds(Qs[wave], DH);

  v8f o0 = {}, o1 = {};
  float mrow[8], lrow[8];
#pragma unroll
  for (int v = 0; v < 8; ++v) { mrow[v] = -3.0e38f; lrow[v] = 0.f; }

  constexpr int NCH = NPTS / 32;
  for (int ch = 0; ch < NCH; ++ch) {
    const int cur = ch & 1;
    if (ch + 1 < NCH) {  // prefetch next chunk into the other buffer
      attn_issue_kv(base, (ch + 1) * 32, h, tid, Ks[1 - cur], Vs[1 - cur]);
      wait_async<2>();   // current chunk complete; next may stay in flight
    } else {
      wait_async<0>();
    }
    __syncthreads();

    v16bf bk0 = load_frag_a_lds(Ks[cur], DH);  // B^T of [key][d] == A-load
    v16bf bk1 = load_frag_a_lds(Ks[cur] + 16 * DH, DH);
    v8f z = {};
    v8f s0 = wmma_bf16(qa, bk0, z);
    v8f s1 = wmma_bf16(qa, bk1, z);

#pragma unroll
    for (int v = 0; v < 8; ++v) {
      const float sv0 = s0[v] * scale;
      const float sv1 = s1[v] * scale;
      float cm = fmaxf(sv0, sv1);
#pragma unroll
      for (int msk = 1; msk < 16; msk <<= 1)
        cm = fmaxf(cm, __shfl_xor(cm, msk, 32));
      const float nm = fmaxf(mrow[v], cm);
      const float alpha = __expf(mrow[v] - nm);
      const float p0 = __expf(sv0 - nm);
      const float p1 = __expf(sv1 - nm);
      float rs = p0 + p1;
#pragma unroll
      for (int msk = 1; msk < 16; msk <<= 1)
        rs += __shfl_xor(rs, msk, 32);
      lrow[v] = lrow[v] * alpha + rs;
      mrow[v] = nm;
      o0[v] *= alpha;
      o1[v] *= alpha;
      Ps[wave][(v + 8 * g) * 32 + (lane & 15)]      = (bf16_t)p0;
      Ps[wave][(v + 8 * g) * 32 + 16 + (lane & 15)] = (bf16_t)p1;
    }
    __syncthreads();

    v16bf pa  = load_frag_a_lds(Ps[wave], 32);
    v16bf bv0 = load_frag_b_lds(Vs[cur], DH);
    v16bf bv1 = load_frag_b_lds(Vs[cur] + 16, DH);
    o0 = wmma_bf16(pa, bv0, o0);
    o1 = wmma_bf16(pa, bv1, o1);
    __syncthreads();  // protect buffer reuse two iterations out
  }

#pragma unroll
  for (int v = 0; v < 8; ++v) {
    const int m = n0 + v + 8 * g;
    const float inv = 1.f / lrow[v];
    att[((size_t)b * NPTS + m) * 128 + h * DH + (lane & 15)]      = (bf16_t)(o0[v] * inv);
    att[((size_t)b * NPTS + m) * 128 + h * DH + 16 + (lane & 15)] = (bf16_t)(o1[v] * inv);
  }
}

// ---------------------------------------------------------------------------
// 6) concat [local | glob] into comb cols 0..255 (out-proj fills 256..383)
// ---------------------------------------------------------------------------
__global__ void concat_kernel(const bf16_t* __restrict__ local,
                              const float* __restrict__ glob,
                              bf16_t* __restrict__ comb) {
  const size_t t = (size_t)blockIdx.x * 256 + threadIdx.x;
  if (t >= (size_t)BATCH * NPTS * 256) return;
  const size_t row = t >> 8;
  const int c = (int)(t & 255);
  const int b = (int)(row / NPTS);
  comb[row * 384 + c] =
      (c < 128) ? local[row * 128 + c] : (bf16_t)glob[b * 128 + (c - 128)];
}

// ---------------------------------------------------------------------------
// 7) in-place log-softmax over 20 logits per row
// ---------------------------------------------------------------------------
__global__ void logsoftmax_kernel(float* __restrict__ logits) {
  const int r = blockIdx.x * 256 + threadIdx.x;
  if (r >= BATCH * NPTS) return;
  float* p = logits + (size_t)r * 20;
  float m = -3.0e38f;
  for (int i = 0; i < 20; ++i) m = fmaxf(m, p[i]);
  float s = 0.f;
  for (int i = 0; i < 20; ++i) s += __expf(p[i] - m);
  const float lse = m + __logf(s);
  for (int i = 0; i < 20; ++i) p[i] = p[i] - lse;
}

// ---------------------------------------------------------------------------
extern "C" void kernel_launch(void* const* d_in, const int* in_sizes, int n_in,
                              void* d_out, int out_size, void* d_ws,
                              size_t ws_size, hipStream_t stream) {
  const float* x      = (const float*)d_in[0];
  const float* ec1_w1 = (const float*)d_in[2];
  const float* ec1_b1 = (const float*)d_in[3];
  const float* ec1_w2 = (const float*)d_in[4];
  const float* ec1_b2 = (const float*)d_in[5];
  const float* ec2_w1 = (const float*)d_in[6];
  const float* ec2_b1 = (const float*)d_in[7];
  const float* ec2_w2 = (const float*)d_in[8];
  const float* ec2_b2 = (const float*)d_in[9];
  const float* qkv_w  = (const float*)d_in[10];
  const float* out_w  = (const float*)d_in[11];
  const float* out_b  = (const float*)d_in[12];
  const float* r1_w   = (const float*)d_in[13];
  const float* r1_b   = (const float*)d_in[14];
  const float* r2_w   = (const float*)d_in[15];
  const float* r2_b   = (const float*)d_in[16];
  const float* r3_w   = (const float*)d_in[17];
  const float* r3_b   = (const float*)d_in[18];

  const size_t R = (size_t)BATCH * NPTS;  // 16384 rows
  char* ws = (char*)d_ws;
  auto alloc = [&](size_t bytes) {
    char* p = ws;
    ws += (bytes + 255) & ~(size_t)255;
    return p;
  };
  int*    idx_ws = (int*)   alloc(R * KNN * sizeof(int));
  float*  feat1  = (float*) alloc(R * 64  * sizeof(float));
  bf16_t* feat2  = (bf16_t*)alloc(R * 128 * sizeof(bf16_t));
  float*  globv  = (float*) alloc((size_t)BATCH * 128 * sizeof(float));
  bf16_t* qkvb   = (bf16_t*)alloc(R * 384 * sizeof(bf16_t));
  bf16_t* attraw = (bf16_t*)alloc(R * 128 * sizeof(bf16_t));
  bf16_t* comb   = (bf16_t*)alloc(R * 384 * sizeof(bf16_t));
  bf16_t* h1     = (bf16_t*)alloc(R * 128 * sizeof(bf16_t));
  bf16_t* h2     = (bf16_t*)alloc(R * 64  * sizeof(bf16_t));
  bf16_t* wq     = (bf16_t*)alloc(128 * 384 * sizeof(bf16_t));
  bf16_t* wo     = (bf16_t*)alloc(128 * 128 * sizeof(bf16_t));
  bf16_t* wr1    = (bf16_t*)alloc(384 * 128 * sizeof(bf16_t));
  bf16_t* wr2    = (bf16_t*)alloc(128 * 64  * sizeof(bf16_t));
  bf16_t* wr3    = (bf16_t*)alloc(64  * 20  * sizeof(bf16_t));
  float*  logits = (float*)d_out;  // [16384 x 20]

  const dim3 blk128(128);
  auto cvt = [&](const float* s, bf16_t* d, int n) {
    cvt_bf16_kernel<<<dim3((n + 255) / 256), dim3(256), 0, stream>>>(s, d, n);
  };
  cvt(qkv_w, wq, 128 * 384);
  cvt(out_w, wo, 128 * 128);
  cvt(r1_w, wr1, 384 * 128);
  cvt(r2_w, wr2, 128 * 64);
  cvt(r3_w, wr3, 64 * 20);

  // EdgeConv 1 (xyz -> 64, f32 out for the feature-space kNN)
  knn_kernel<3><<<dim3(NPTS / 128, BATCH), blk128, 0, stream>>>(x, idx_ws);
  edge_conv_kernel<3, 64, float><<<dim3(NPTS / 4, BATCH), blk128, 0, stream>>>(
      x, idx_ws, ec1_w1, ec1_b1, ec1_w2, ec1_b2, feat1);

  // EdgeConv 2 (64 -> 128, bf16 out feeds all downstream WMMA consumers)
  knn_kernel<64><<<dim3(NPTS / 128, BATCH), blk128, 0, stream>>>(feat1, idx_ws);
  edge_conv_kernel<64, 128, bf16_t><<<dim3(NPTS / 4, BATCH), blk128, 0, stream>>>(
      feat1, idx_ws, ec2_w1, ec2_b1, ec2_w2, ec2_b2, feat2);

  // global max pool
  global_max_kernel<<<dim3(BATCH), blk128, 0, stream>>>(feat2, globv);

  // attention: qkv projection (bf16), flash attention, out-proj into comb[:,256:]
  gemm_kernel<0, bf16_t><<<dim3((unsigned)(R / 64), 6), blk128, 0, stream>>>(
      feat2, 128, wq, 384, nullptr, qkvb, 384, 0, 128, 384, 1);
  attention_kernel<<<dim3(NPTS / 64, 4, BATCH), blk128, 0, stream>>>(qkvb, attraw);

  concat_kernel<<<dim3((unsigned)((R * 256 + 255) / 256)), dim3(256), 0, stream>>>(
      feat2, globv, comb);
  gemm_kernel<0, bf16_t><<<dim3((unsigned)(R / 64), 2), blk128, 0, stream>>>(
      attraw, 128, wo, 128, out_b, comb, 384, 256, 128, 128, 1);

  // refine MLP 384 -> 128 -> 64 -> 20, then log-softmax in place on d_out
  gemm_kernel<1, bf16_t><<<dim3((unsigned)(R / 64), 2), blk128, 0, stream>>>(
      comb, 384, wr1, 128, r1_b, h1, 128, 0, 384, 128, 1);
  gemm_kernel<1, bf16_t><<<dim3((unsigned)(R / 64), 1), blk128, 0, stream>>>(
      h1, 128, wr2, 64, r2_b, h2, 64, 0, 128, 64, 1);
  gemm_kernel<0, float><<<dim3((unsigned)(R / 64), 1), blk128, 0, stream>>>(
      h2, 64, wr3, 20, r3_b, logits, 20, 0, 64, 20, 0);
  logsoftmax_kernel<<<dim3((unsigned)((R + 255) / 256)), dim3(256), 0, stream>>>(logits);
}
